// SelfAttention_22565758174057
// MI455X (gfx1250) — compile-verified, matrix operand order checked by reading, not run on hardware
//
#include <hip/hip_runtime.h>
#include <math.h>

typedef __attribute__((ext_vector_type(16))) __bf16 v16bf;
typedef __attribute__((ext_vector_type(8)))  float  v8f;
typedef __bf16 bf16_t;

#define HD   128
#define B_   2
#define S_   2048
#define D_   4096
#define KVD_ 1024
#define H_   32
#define KVH_ 8

// ---- WMMA helpers ----------------------------------------------------------

static __device__ __forceinline__ v8f wmma_bf16(v16bf a, v16bf b, v8f c) {
  // D = A(16x32 bf16) * B(32x16 bf16) + C(16x16 f32)
  return __builtin_amdgcn_wmma_f32_16x16x32_bf16(false, a, false, b,
                                                 (short)0, c, false, false);
}

// Load a 16x32 bf16 fragment (A layout; also used for B via transposed tiles).
// Lane l: row = l%16; half = l/16; VGPR v<4 -> K = 2v+8*half (+1),
// v>=4 -> K = 16+2(v-4)+8*half (+1).   (ISA 7.12.2, 16-bit A 16x32)
// Per lane this is two contiguous 16B spans -> 2 x b128 loads.
static __device__ __forceinline__ v16bf load_frag16x32(const bf16_t* base,
                                                       int rowStride) {
  const int l = threadIdx.x & 31;
  const int half = (l >> 4) & 1;
  const bf16_t* row = base + (size_t)(l & 15) * rowStride;
  v16bf f;
#pragma unroll
  for (int v = 0; v < 8; ++v) {
    const int k = ((v < 4) ? (2 * v) : (2 * v + 8)) + 8 * half;
    f[2 * v]     = row[k];
    f[2 * v + 1] = row[k + 1];
  }
  return f;
}

// ---- Kernel 1: fused QKV projection (fp32 x, fp32 W -> bf16 Q/K/V^T) -------

__global__ __launch_bounds__(256)
void qkv_gemm_kernel(const float* __restrict__ x,
                     const float* __restrict__ wq,
                     const float* __restrict__ wk,
                     const float* __restrict__ wv,
                     bf16_t* __restrict__ qbuf,   // [B,H,S,HD]
                     bf16_t* __restrict__ kbuf,   // [B,KVH,S,HD]
                     bf16_t* __restrict__ vT) {   // [B,KVH,HD,S]
  __shared__ bf16_t As[128 * 32];
  __shared__ bf16_t BsT[128 * 32];

  const int t    = threadIdx.x;
  const int lane = t & 31;
  const int wid  = t >> 5;
  const int nTile = blockIdx.x * 128;   // 0..6143
  const int mTile = blockIdx.y * 128;   // 0..4095 (rows = B*S)

  const float* Wp; int ldw, ncol, mode;
  if (nTile < D_)              { Wp = wq; ldw = D_;   ncol = nTile;              mode = 0; }
  else if (nTile < D_ + KVD_)  { Wp = wk; ldw = KVD_; ncol = nTile - D_;         mode = 1; }
  else                         { Wp = wv; ldw = KVD_; ncol = nTile - D_ - KVD_;  mode = 2; }

  const int waveM = wid & 3;   // 4 waves along M (x32)
  const int waveN = wid >> 2;  // 2 waves along N (x64)

  const v8f vzero = {0.f,0.f,0.f,0.f,0.f,0.f,0.f,0.f};
  v8f acc[2][4];
#pragma unroll
  for (int i = 0; i < 2; ++i)
#pragma unroll
    for (int j = 0; j < 4; ++j) acc[i][j] = vzero;

  // register staging for the global->LDS pipeline
  float4 aR[4], wR[4];

  auto issue_loads = [&](int kt) {
#pragma unroll
    for (int i = 0; i < 4; ++i) {
      const int vi = t + i * 256;      // float4 slot
      const int row = vi >> 3;         // 8 float4 per 32-col A row
      const int c4  = (vi & 7) << 2;
      aR[i] = *(const float4*)(x + (size_t)(mTile + row) * D_ + kt + c4);
    }
#pragma unroll
    for (int i = 0; i < 4; ++i) {
      const int vi = t + i * 256;
      const int row = vi >> 5;         // 0..31
      const int c4  = (vi & 31) << 2;  // 0..124
      wR[i] = *(const float4*)(Wp + (size_t)(kt + row) * ldw + ncol + c4);
    }
  };

  auto commit_tiles = [&]() {
#pragma unroll
    for (int i = 0; i < 4; ++i) {
      const int vi = t + i * 256;
      const int row = vi >> 3;
      const int c4  = (vi & 7) << 2;
      bf16_t* dst = As + row * 32 + c4;
      dst[0] = (bf16_t)aR[i].x; dst[1] = (bf16_t)aR[i].y;
      dst[2] = (bf16_t)aR[i].z; dst[3] = (bf16_t)aR[i].w;
    }
#pragma unroll
    for (int i = 0; i < 4; ++i) {
      const int vi = t + i * 256;
      const int row = vi >> 5;
      const int c4  = (vi & 31) << 2;
      BsT[(c4 + 0) * 32 + row] = (bf16_t)wR[i].x;
      BsT[(c4 + 1) * 32 + row] = (bf16_t)wR[i].y;
      BsT[(c4 + 2) * 32 + row] = (bf16_t)wR[i].z;
      BsT[(c4 + 3) * 32 + row] = (bf16_t)wR[i].w;
    }
  };

  issue_loads(0);
  for (int kt = 0; kt < D_; kt += 32) {
    commit_tiles();                 // one loadcnt wait, then LDS stores
    __syncthreads();
    if (kt + 32 < D_) issue_loads(kt + 32);  // overlap with WMMA below

    v16bf af[2], bfr[4];
#pragma unroll
    for (int i = 0; i < 2; ++i)
      af[i] = load_frag16x32(As + (waveM * 32 + i * 16) * 32, 32);
#pragma unroll
    for (int j = 0; j < 4; ++j)
      bfr[j] = load_frag16x32(BsT + (waveN * 64 + j * 16) * 32, 32);
#pragma unroll
    for (int i = 0; i < 2; ++i)
#pragma unroll
      for (int j = 0; j < 4; ++j)
        acc[i][j] = wmma_bf16(af[i], bfr[j], acc[i][j]);
    __syncthreads();
  }

  // epilogue: scatter to Q / K / V^T (a 128-wide tile never straddles matrices)
  const int half = (lane >> 4) & 1;
#pragma unroll
  for (int i = 0; i < 2; ++i) {
#pragma unroll
    for (int j = 0; j < 4; ++j) {
#pragma unroll
      for (int r = 0; r < 8; ++r) {
        const int row = mTile + waveM * 32 + i * 16 + r + 8 * half;
        const int gc  = nTile + waveN * 64 + j * 16 + (lane & 15);
        const int b = row >> 11, s = row & (S_ - 1);
        const bf16_t val = (bf16_t)acc[i][j][r];
        if (mode == 0) {
          const int h = gc >> 7, d = gc & 127;
          qbuf[(((size_t)(b * H_ + h)) * S_ + s) * HD + d] = val;
        } else if (mode == 1) {
          const int c = gc - D_, g = c >> 7, d = c & 127;
          kbuf[(((size_t)(b * KVH_ + g)) * S_ + s) * HD + d] = val;
        } else {
          const int c = gc - D_ - KVD_, g = c >> 7, d = c & 127;
          vT[(((size_t)(b * KVH_ + g)) * HD + d) * S_ + s] = val;
        }
      }
    }
  }
}

// ---- Kernel 2: RoPE on Q and K (in-place, bf16) ----------------------------

__global__ __launch_bounds__(256)
void rope_kernel(bf16_t* __restrict__ qbuf, bf16_t* __restrict__ kbuf,
                 const float* __restrict__ cosb, const float* __restrict__ sinb) {
  const long long qPairs = (long long)B_ * H_ * S_ * (HD / 2);
  const long long kPairs = (long long)B_ * KVH_ * S_ * (HD / 2);
  long long idx = (long long)blockIdx.x * blockDim.x + threadIdx.x;
  bf16_t* p; int s, i;
  if (idx < qPairs) {
    i = (int)(idx & 63);
    long long rowi = idx >> 6;           // (b*H+h)*S + s
    s = (int)(rowi & (S_ - 1));
    p = qbuf + (rowi << 7) + 2 * i;
  } else if (idx < qPairs + kPairs) {
    long long kidx = idx - qPairs;
    i = (int)(kidx & 63);
    long long rowi = kidx >> 6;
    s = (int)(rowi & (S_ - 1));
    p = kbuf + (rowi << 7) + 2 * i;
  } else {
    return;
  }
  const float c  = cosb[s * 64 + i];
  const float sn = sinb[s * 64 + i];
  const float re = (float)p[0];
  const float im = (float)p[1];
  p[0] = (bf16_t)(re * c - im * sn);
  p[1] = (bf16_t)(re * sn + im * c);
}

// ---- Kernel 3: flash attention (GQA, non-causal) ---------------------------

__global__ __launch_bounds__(128)
void attn_kernel(const bf16_t* __restrict__ qbuf,
                 const bf16_t* __restrict__ kbuf,
                 const bf16_t* __restrict__ vT,
                 bf16_t* __restrict__ attn) {       // [B,S,H*HD]
  __shared__ bf16_t Plds[4][16 * 32];

  const int lane = threadIdx.x & 31;
  const int wvid = threadIdx.x >> 5;   // 0..3
  const int half = (lane >> 4) & 1;

  const int qb = blockIdx.x;           // 0..S/64-1
  const int h  = blockIdx.y;           // 0..31
  const int b  = blockIdx.z;           // 0..1
  const int g  = h >> 2;               // kv head (rep=4)
  const int q0 = qb * 64 + wvid * 16;

  const bf16_t* qhead = qbuf + (((size_t)(b * H_ + h)) * S_ + q0) * HD;
  const bf16_t* khead = kbuf + (((size_t)(b * KVH_ + g)) * S_) * HD;
  const bf16_t* vhead = vT  + (((size_t)(b * KVH_ + g)) * HD) * S_;

  v16bf aq[4];
#pragma unroll
  for (int c = 0; c < 4; ++c) aq[c] = load_frag16x32(qhead + 32 * c, HD);

  const v8f vzero = {0.f,0.f,0.f,0.f,0.f,0.f,0.f,0.f};
  v8f o[8];
#pragma unroll
  for (int j = 0; j < 8; ++j) o[j] = vzero;
  float m_i[8], l_i[8];
#pragma unroll
  for (int r = 0; r < 8; ++r) { m_i[r] = -1e30f; l_i[r] = 0.f; }

  const float scale = 0.08838834764831845f;  // 1/sqrt(128)
  bf16_t* pl = &Plds[wvid][0];

  for (int kb = 0; kb < S_; kb += 32) {
    // --- batch-load all 8 K fragments, then run the 8 score WMMAs ---
    v16bf bk[8];
#pragma unroll
    for (int c = 0; c < 4; ++c) {
      bk[c]     = load_frag16x32(khead + (size_t)kb * HD + 32 * c, HD);
      bk[4 + c] = load_frag16x32(khead + (size_t)(kb + 16) * HD + 32 * c, HD);
    }
    v8f s0 = vzero, s1 = vzero;
#pragma unroll
    for (int c = 0; c < 4; ++c) {
      s0 = wmma_bf16(aq[c], bk[c],     s0);
      s1 = wmma_bf16(aq[c], bk[4 + c], s1);
    }

    // --- issue V fragment loads now; their latency hides under softmax ---
    v16bf bv[8];
#pragma unroll
    for (int j = 0; j < 8; ++j)
      bv[j] = load_frag16x32(vhead + (size_t)(16 * j) * S_ + kb, S_);

    if (kb + 32 < S_)
      __builtin_prefetch(khead + (size_t)(kb + 32) * HD + lane * 4, 0, 1);

    // --- online softmax per row (row m = r + 8*half; N across 16 lanes) ---
#pragma unroll
    for (int r = 0; r < 8; ++r) {
      float x0 = s0[r] * scale;
      float x1 = s1[r] * scale;
      float mx = fmaxf(x0, x1);
#pragma unroll
      for (int d = 1; d < 16; d <<= 1) mx = fmaxf(mx, __shfl_xor(mx, d, 32));
      const float mnew  = fmaxf(m_i[r], mx);
      const float alpha = __expf(m_i[r] - mnew);
      const float p0 = __expf(x0 - mnew);
      const float p1 = __expf(x1 - mnew);
      float rs = p0 + p1;
#pragma unroll
      for (int d = 1; d < 16; d <<= 1) rs += __shfl_xor(rs, d, 32);
      l_i[r] = l_i[r] * alpha + rs;
      m_i[r] = mnew;
#pragma unroll
      for (int j = 0; j < 8; ++j) o[j][r] *= alpha;
      const int m = r + 8 * half;
      pl[m * 32 + (lane & 15)]      = (bf16_t)p0;
      pl[m * 32 + 16 + (lane & 15)] = (bf16_t)p1;
    }
    __syncthreads();

    // --- P(16x32) x V^T chunks(32x16) into 8 output accumulators ---
    const v16bf pf = load_frag16x32(pl, 32);
#pragma unroll
    for (int j = 0; j < 8; ++j) o[j] = wmma_bf16(pf, bv[j], o[j]);
    __syncthreads();
  }

  // normalize + store to attn[b, s, h*128 + d] (bf16)
#pragma unroll
  for (int r = 0; r < 8; ++r) {
    const int s = q0 + r + 8 * half;
    const float inv = 1.f / l_i[r];
#pragma unroll
    for (int j = 0; j < 8; ++j) {
      attn[((size_t)(b * S_ + s)) * D_ + h * HD + j * 16 + (lane & 15)] =
          (bf16_t)(o[j][r] * inv);
    }
  }
}

// ---- Kernel 4: output projection (bf16 attn x fp32 wo -> fp32 out) ---------

__global__ __launch_bounds__(256)
void out_gemm_kernel(const bf16_t* __restrict__ attn,
                     const float* __restrict__ wo,
                     float* __restrict__ out) {
  __shared__ bf16_t As[128 * 32];
  __shared__ bf16_t BsT[128 * 32];

  const int t    = threadIdx.x;
  const int lane = t & 31;
  const int wid  = t >> 5;
  const int nTile = blockIdx.x * 128;
  const int mTile = blockIdx.y * 128;
  const int waveM = wid & 3;
  const int waveN = wid >> 2;

  const v8f vzero = {0.f,0.f,0.f,0.f,0.f,0.f,0.f,0.f};
  v8f acc[2][4];
#pragma unroll
  for (int i = 0; i < 2; ++i)
#pragma unroll
    for (int j = 0; j < 4; ++j) acc[i][j] = vzero;

  uint4  aR[2];
  float4 wR[4];

  auto issue_loads = [&](int kt) {
#pragma unroll
    for (int i = 0; i < 2; ++i) {
      const int vi = t + i * 256;      // 0..511 (8-bf16 slots)
      const int row = vi >> 2;         // 4 slots per 32-col row
      const int c8  = (vi & 3) << 3;
      aR[i] = *(const uint4*)(attn + (size_t)(mTile + row) * D_ + kt + c8);
    }
#pragma unroll
    for (int i = 0; i < 4; ++i) {
      const int vi = t + i * 256;
      const int row = vi >> 5;
      const int c4  = (vi & 31) << 2;
      wR[i] = *(const float4*)(wo + (size_t)(kt + row) * D_ + nTile + c4);
    }
  };

  auto commit_tiles = [&]() {
#pragma unroll
    for (int i = 0; i < 2; ++i) {
      const int vi = t + i * 256;
      const int row = vi >> 2;
      const int c8  = (vi & 3) << 3;
      *(uint4*)(As + row * 32 + c8) = aR[i];
    }
#pragma unroll
    for (int i = 0; i < 4; ++i) {
      const int vi = t + i * 256;
      const int row = vi >> 5;
      const int c4  = (vi & 31) << 2;
      BsT[(c4 + 0) * 32 + row] = (bf16_t)wR[i].x;
      BsT[(c4 + 1) * 32 + row] = (bf16_t)wR[i].y;
      BsT[(c4 + 2) * 32 + row] = (bf16_t)wR[i].z;
      BsT[(c4 + 3) * 32 + row] = (bf16_t)wR[i].w;
    }
  };

  issue_loads(0);
  for (int kt = 0; kt < D_; kt += 32) {
    commit_tiles();
    __syncthreads();
    if (kt + 32 < D_) issue_loads(kt + 32);

    v16bf af[2], bfr[4];
#pragma unroll
    for (int i = 0; i < 2; ++i)
      af[i] = load_frag16x32(As + (waveM * 32 + i * 16) * 32, 32);
#pragma unroll
    for (int j = 0; j < 4; ++j)
      bfr[j] = load_frag16x32(BsT + (waveN * 64 + j * 16) * 32, 32);
#pragma unroll
    for (int i = 0; i < 2; ++i)
#pragma unroll
      for (int j = 0; j < 4; ++j)
        acc[i][j] = wmma_bf16(af[i], bfr[j], acc[i][j]);
    __syncthreads();
  }

  const int half = (lane >> 4) & 1;
#pragma unroll
  for (int i = 0; i < 2; ++i)
#pragma unroll
    for (int j = 0; j < 4; ++j)
#pragma unroll
      for (int r = 0; r < 8; ++r) {
        const int row = mTile + waveM * 32 + i * 16 + r + 8 * half;
        const int col = nTile + waveN * 64 + j * 16 + (lane & 15);
        out[(size_t)row * D_ + col] = acc[i][j][r];
      }
}

// ---- Host launch -----------------------------------------------------------

extern "C" void kernel_launch(void* const* d_in, const int* in_sizes, int n_in,
                              void* d_out, int out_size, void* d_ws, size_t ws_size,
                              hipStream_t stream) {
  const float* x    = (const float*)d_in[0];
  const float* cosb = (const float*)d_in[1];
  const float* sinb = (const float*)d_in[2];
  const float* wq   = (const float*)d_in[3];
  const float* wk   = (const float*)d_in[4];
  const float* wv   = (const float*)d_in[5];
  const float* wo   = (const float*)d_in[6];
  float* out = (float*)d_out;

  bf16_t* qbuf = (bf16_t*)d_ws;                               // 16Mi elems
  bf16_t* kbuf = qbuf + (size_t)B_ * H_ * S_ * HD;            // 4Mi elems
  bf16_t* vT   = kbuf + (size_t)B_ * KVH_ * S_ * HD;          // 4Mi elems
  bf16_t* attn = vT   + (size_t)B_ * KVH_ * S_ * HD;          // 16Mi elems

  dim3 g1((D_ + 2 * KVD_) / 128, (B_ * S_) / 128);            // 48 x 32
  qkv_gemm_kernel<<<g1, 256, 0, stream>>>(x, wq, wk, wv, qbuf, kbuf, vT);

  const long long pairs = (long long)B_ * H_ * S_ * 64 +
                          (long long)B_ * KVH_ * S_ * 64;
  rope_kernel<<<(int)((pairs + 255) / 256), 256, 0, stream>>>(qbuf, kbuf,
                                                              cosb, sinb);

  attn_kernel<<<dim3(S_ / 64, H_, B_), 128, 0, stream>>>(qbuf, kbuf, vT, attn);

  out_gemm_kernel<<<dim3(D_ / 128, (B_ * S_) / 128), 256, 0, stream>>>(attn, wo,
                                                                       out);
}
